// _BaseGraphWrapper_31739808318050
// MI455X (gfx1250) — matrix operand order repaired and checked by reading, not compiled, not run on hardware
//
#include <hip/hip_runtime.h>
#include <hip/hip_bf16.h>
#include <math.h>

typedef _Float16 f16;
typedef __attribute__((ext_vector_type(16))) _Float16 v16h;
typedef __attribute__((ext_vector_type(8)))  _Float16 v8h;
typedef __attribute__((ext_vector_type(8)))  float    v8f;

namespace {
constexpr int kB   = 32;    // graphs
constexpr int kN   = 512;   // nodes per graph
constexpr int kE   = 8192;  // edges per graph
constexpr int kIn  = 64;
constexpr int kD   = 128;
constexpr int kH   = 4;
constexpr int kHD  = 32;
constexpr int kL   = 3;
constexpr int kC   = 10;
constexpr int kHid = 256;
constexpr int kBN  = kB * kN;   // 16384 nodes total
constexpr int kBE  = kB * kE;   // 262144 edges total
}

union F16Frag { v16h v; v8h p[2]; };

__device__ __forceinline__ v8f zero8() {
  v8f z;
#pragma unroll
  for (int i = 0; i < 8; ++i) z[i] = 0.f;
  return z;
}

__device__ __forceinline__ float gelu_f(float x) {
  return 0.5f * x * (1.f + erff(x * 0.70710678118654752f));
}

// A-matrix (16xK panel, row-major f16, strideH halfs): CDNA5 16-bit A layout.
// lanes 0-15: row=lane, K chunks {kb+0..7, kb+16..23}; lanes 16-31: row=lane-16,
// K chunks {kb+8..15, kb+24..31}.
__device__ __forceinline__ v16h load_a_frag(const f16* a0, int strideH, int kb, int lane) {
  F16Frag u;
  int row = lane & 15, hi = lane >> 4;
  const f16* p = a0 + (size_t)row * strideH + kb + hi * 8;
  u.p[0] = *(const v8h*)p;
  u.p[1] = *(const v8h*)(p + 16);
  return u.v;
}

// B-matrix from column-major [N][K] f16 (strideH halfs per column):
// lanes 0-15: col=lane, K chunks {kb+0..7, kb+16..23}; lanes 16-31: col=lane-16, K chunks {kb+8..15, kb+24..31}.
__device__ __forceinline__ v16h load_b_frag(const f16* bT, int strideH, int nb, int kb, int lane) {
  F16Frag u;
  int col = lane & 15, hi = lane >> 4;
  const f16* p = bT + (size_t)(nb + col) * strideH + kb + hi * 8;
  u.p[0] = *(const v8h*)p;
  u.p[1] = *(const v8h*)(p + 16);
  return u.v;
}

__device__ __forceinline__ v8f wmma_f16(v16h a, v16h b, v8f c) {
  return __builtin_amdgcn_wmma_f32_16x16x32_f16(false, a, false, b, (short)0, c, false, false);
}

// ---------------- pack kernels ----------------

__global__ void k_pack_wT(const float* __restrict__ w, f16* __restrict__ wT, int K, int Nc) {
  int total = K * Nc;
  for (int idx = blockIdx.x * blockDim.x + threadIdx.x; idx < total; idx += gridDim.x * blockDim.x) {
    int k2 = idx / Nc, n = idx % Nc;          // coalesced read of w[k][n]
    wT[(size_t)n * K + k2] = (f16)w[idx];     // column-major store
  }
}

__global__ void k_pack_f16(const float* __restrict__ a, f16* __restrict__ ap, int total) {
  for (int i = blockIdx.x * blockDim.x + threadIdx.x; i < total; i += gridDim.x * blockDim.x)
    ap[i] = (f16)a[i];
}

// ---------------- node encoder ----------------

// u = gelu(LN(xp @ W1 + b1))  ; xp f16[kBN][64], W1T f16[256][64], up f16[kBN][256]
__global__ __launch_bounds__(256) void k_enc1(const f16* __restrict__ xp, const f16* __restrict__ w1T,
                                              const float* __restrict__ b1, const float* __restrict__ g,
                                              const float* __restrict__ bb, f16* __restrict__ up) {
  __shared__ float tb[16][kHid];
  __shared__ float red0[16][16], red1[16][16];
  __shared__ float smu[16], srs[16];
  int mt = blockIdx.x, wv = threadIdx.x >> 5, lane = threadIdx.x & 31;
  const f16* A = xp + (size_t)mt * 16 * kIn;
  for (int nt = wv * 2; nt < wv * 2 + 2; ++nt) {
    v8f acc = zero8();
#pragma unroll
    for (int kb = 0; kb < kIn; kb += 32)
      acc = wmma_f16(load_a_frag(A, kIn, kb, lane), load_b_frag(w1T, kIn, nt * 16, kb, lane), acc);
    int hi = lane >> 4, col = nt * 16 + (lane & 15);
#pragma unroll
    for (int i = 0; i < 8; ++i) tb[i + 8 * hi][col] = acc[i] + b1[col];
  }
  __syncthreads();
  int row = threadIdx.x >> 4, seg = threadIdx.x & 15;
  float s = 0.f, s2 = 0.f;
#pragma unroll
  for (int j = 0; j < 16; ++j) { float v = tb[row][seg * 16 + j]; s += v; s2 += v * v; }
  red0[row][seg] = s; red1[row][seg] = s2;
  __syncthreads();
  if (seg == 0) {
    float ts = 0.f, ts2 = 0.f;
#pragma unroll
    for (int j = 0; j < 16; ++j) { ts += red0[row][j]; ts2 += red1[row][j]; }
    float m = ts / kHid;
    smu[row] = m;
    srs[row] = rsqrtf(ts2 / kHid - m * m + 1e-5f);
  }
  __syncthreads();
#pragma unroll
  for (int j = 0; j < 16; ++j) {
    int c = seg * 16 + j;
    float v = (tb[row][c] - smu[row]) * srs[row] * g[c] + bb[c];
    up[(size_t)(mt * 16 + row) * kHid + c] = (f16)gelu_f(v);
  }
}

// h = LN(up @ W2 + b2)  ; up f16[kBN][256], W2T f16[128][256] -> h f32, hp f16
__global__ __launch_bounds__(256) void k_enc2(const f16* __restrict__ up, const f16* __restrict__ w2T,
                                              const float* __restrict__ b2, const float* __restrict__ g,
                                              const float* __restrict__ bb, float* __restrict__ h,
                                              f16* __restrict__ hp) {
  __shared__ float tb[16][kD];
  __shared__ float red0[16][16], red1[16][16];
  __shared__ float smu[16], srs[16];
  int mt = blockIdx.x, wv = threadIdx.x >> 5, lane = threadIdx.x & 31;
  const f16* A = up + (size_t)mt * 16 * kHid;
  v8f acc = zero8();
#pragma unroll
  for (int kb = 0; kb < kHid; kb += 32)
    acc = wmma_f16(load_a_frag(A, kHid, kb, lane), load_b_frag(w2T, kHid, wv * 16, kb, lane), acc);
  int hi = lane >> 4, col = wv * 16 + (lane & 15);
#pragma unroll
  for (int i = 0; i < 8; ++i) tb[i + 8 * hi][col] = acc[i] + b2[col];
  __syncthreads();
  int row = threadIdx.x >> 4, seg = threadIdx.x & 15;
  float s = 0.f, s2 = 0.f;
#pragma unroll
  for (int j = 0; j < 8; ++j) { float v = tb[row][seg * 8 + j]; s += v; s2 += v * v; }
  red0[row][seg] = s; red1[row][seg] = s2;
  __syncthreads();
  if (seg == 0) {
    float ts = 0.f, ts2 = 0.f;
#pragma unroll
    for (int j = 0; j < 16; ++j) { ts += red0[row][j]; ts2 += red1[row][j]; }
    float m = ts / kD;
    smu[row] = m;
    srs[row] = rsqrtf(ts2 / kD - m * m + 1e-5f);
  }
  __syncthreads();
#pragma unroll
  for (int j = 0; j < 8; ++j) {
    int c = seg * 8 + j;
    float v = (tb[row][c] - smu[row]) * srs[row] * g[c] + bb[c];
    size_t idx = (size_t)(mt * 16 + row) * kD + c;
    h[idx] = v;
    hp[idx] = (f16)v;
  }
}

// ---------------- edge-message MLP + scatter-add ----------------

__global__ __launch_bounds__(256) void k_edge(const f16* __restrict__ hp, const int* __restrict__ src,
                                              const int* __restrict__ dst, const f16* __restrict__ w1T,
                                              const float* __restrict__ b1, const f16* __restrict__ w2T,
                                              const float* __restrict__ b2, float* __restrict__ accb) {
  __shared__ __align__(16) f16 ef[16][264];   // gathered [hs|hd], K=256, padded stride
  __shared__ __align__(16) f16 mid[16][136];  // gelu(GEMM1) f16, K=128, padded stride
  __shared__ int sIdx[16], dIdx[16];
  int wv = threadIdx.x >> 5, lane = threadIdx.x & 31;
  for (int it = 0; it < 4; ++it) {
    int tile = blockIdx.x * 4 + it;
    if (threadIdx.x < 16) {
      int e = tile * 16 + threadIdx.x;
      int b_ = e >> 13, ee = e & (kE - 1);
      sIdx[threadIdx.x] = b_ * kN + src[(size_t)b_ * kE + ee];
      dIdx[threadIdx.x] = b_ * kN + dst[(size_t)b_ * kE + ee];
    } else if (threadIdx.x >= 32 && threadIdx.x < 64 && it < 3) {
      // prefetch next tile's 32 gathered feature rows (256 B each = 2 lines)
      // while this tile's 12 WMMAs execute -> global_prefetch_b8
      int t2 = threadIdx.x - 32;              // 0..31: 16 src rows + 16 dst rows
      int e = (tile + 1) * 16 + (t2 & 15);
      int b_ = e >> 13, ee = e & (kE - 1);
      const int* idxp = (t2 < 16) ? src : dst;
      int node = b_ * kN + idxp[(size_t)b_ * kE + ee];
      const f16* rowp = hp + (size_t)node * kD;
      __builtin_prefetch(rowp, 0, 0);
      __builtin_prefetch(rowp + 64, 0, 0);
    }
    __syncthreads();
    {
      int r = threadIdx.x >> 4, seg = threadIdx.x & 15;
      const f16* srow = hp + (size_t)sIdx[r] * kD;
      const f16* drow = hp + (size_t)dIdx[r] * kD;
#pragma unroll
      for (int j = 0; j < 16; ++j) {
        int c = seg * 16 + j;
        ef[r][c] = (c < kD) ? srow[c] : drow[c - kD];
      }
    }
    __syncthreads();
    v8f acc = zero8();
#pragma unroll
    for (int kb = 0; kb < 2 * kD; kb += 32)
      acc = wmma_f16(load_a_frag(&ef[0][0], 264, kb, lane), load_b_frag(w1T, 2 * kD, wv * 16, kb, lane), acc);
    {
      int hi = lane >> 4, col = wv * 16 + (lane & 15);
#pragma unroll
      for (int i = 0; i < 8; ++i) mid[i + 8 * hi][col] = (f16)gelu_f(acc[i] + b1[col]);
    }
    __syncthreads();
    v8f a2 = zero8();
#pragma unroll
    for (int kb = 0; kb < kD; kb += 32)
      a2 = wmma_f16(load_a_frag(&mid[0][0], 136, kb, lane), load_b_frag(w2T, kD, wv * 16, kb, lane), a2);
    {
      int hi = lane >> 4, col = wv * 16 + (lane & 15);
#pragma unroll
      for (int i = 0; i < 8; ++i) {
        int er = i + 8 * hi;
        atomicAdd(&accb[(size_t)dIdx[er] * kD + col], a2[i] + b2[col]);
      }
    }
    __syncthreads();
  }
}

// h = LN(h + acc) * g + b ; refresh hp
__global__ __launch_bounds__(128) void k_resln(float* __restrict__ h, const float* __restrict__ accb,
                                               const float* __restrict__ g, const float* __restrict__ bb,
                                               f16* __restrict__ hp) {
  __shared__ float r0[128], r1[128];
  int row = blockIdx.x, t = threadIdx.x;
  size_t idx = (size_t)row * kD + t;
  float v = h[idx] + accb[idx];
  r0[t] = v; r1[t] = v * v;
  __syncthreads();
  for (int off = 64; off; off >>= 1) {
    if (t < off) { r0[t] += r0[t + off]; r1[t] += r1[t + off]; }
    __syncthreads();
  }
  float m = r0[0] / kD;
  float rs = rsqrtf(r1[0] / kD - m * m + 1e-5f);
  float o = (v - m) * rs * g[t] + bb[t];
  h[idx] = o;
  hp[idx] = (f16)o;
}

// ---------------- attention ----------------

// MODE 0: write [b][h][n][32] (q,k A/B layout). MODE 1: write [b][h][32][512] (v^T).
template <int MODE>
__global__ __launch_bounds__(256) void k_proj(const f16* __restrict__ hp, const f16* __restrict__ wT,
                                              const float* __restrict__ bias, f16* __restrict__ outp) {
  int mt = blockIdx.x, wv = threadIdx.x >> 5, lane = threadIdx.x & 31;
  const f16* A = hp + (size_t)mt * 16 * kD;
  v8f acc = zero8();
#pragma unroll
  for (int kb = 0; kb < kD; kb += 32)
    acc = wmma_f16(load_a_frag(A, kD, kb, lane), load_b_frag(wT, kD, wv * 16, kb, lane), acc);
  int hi = lane >> 4, col = wv * 16 + (lane & 15);
  int hh = col >> 5, d = col & 31;
#pragma unroll
  for (int i = 0; i < 8; ++i) {
    int grow = mt * 16 + i + 8 * hi;
    int b_ = grow >> 9, n = grow & (kN - 1);
    float v = acc[i] + bias[col];
    if (MODE == 0) outp[(((size_t)b_ * kH + hh) * kN + n) * kHD + d] = (f16)v;
    else           outp[(((size_t)b_ * kH + hh) * kHD + d) * kN + n] = (f16)v;
  }
}

// S[bh][n][m] = (q . k) / sqrt(32) ; one wmma per 16x16 tile
__global__ __launch_bounds__(256) void k_scores(const f16* __restrict__ qp, const f16* __restrict__ kp,
                                                f16* __restrict__ S) {
  int task = blockIdx.x * 8 + (threadIdx.x >> 5);
  int lane = threadIdx.x & 31;
  int bh = task >> 10, rem = task & 1023;
  int mt = rem >> 5, nt = rem & 31;
  const f16* A = qp + (size_t)bh * kN * kHD + (size_t)mt * 16 * kHD;
  v16h a = load_a_frag(A, kHD, 0, lane);
  v16h b = load_b_frag(kp + (size_t)bh * kN * kHD, kHD, nt * 16, 0, lane);
  v8f acc = wmma_f16(a, b, zero8());
  int hi = lane >> 4, col = nt * 16 + (lane & 15);
  f16* Sp = S + (size_t)bh * kN * kN;
  const float scale = 0.17677669529663687f;  // 1/sqrt(32)
#pragma unroll
  for (int i = 0; i < 8; ++i)
    Sp[(size_t)(mt * 16 + i + 8 * hi) * kN + col] = (f16)(acc[i] * scale);
}

// in-place row softmax over 512, one wave per row
__global__ __launch_bounds__(256) void k_softmax(f16* __restrict__ S) {
  int row = blockIdx.x * 8 + (threadIdx.x >> 5);
  int lane = threadIdx.x & 31;
  f16* p = S + (size_t)row * kN;
  float vals[16];
  float mx = -1e30f;
#pragma unroll
  for (int i = 0; i < 16; ++i) { float v = (float)p[lane + 32 * i]; vals[i] = v; mx = fmaxf(mx, v); }
  for (int off = 16; off; off >>= 1) mx = fmaxf(mx, __shfl_xor(mx, off, 32));
  float s = 0.f;
#pragma unroll
  for (int i = 0; i < 16; ++i) { float e = expf(vals[i] - mx); vals[i] = e; s += e; }
  for (int off = 16; off; off >>= 1) s += __shfl_xor(s, off, 32);
  float inv = 1.f / s;
#pragma unroll
  for (int i = 0; i < 16; ++i) p[lane + 32 * i] = (f16)(vals[i] * inv);
}

// O = P @ V ; writes f16 op[kBN][128] for the output projection
__global__ __launch_bounds__(256) void k_av(const f16* __restrict__ S, const f16* __restrict__ vp,
                                            f16* __restrict__ op) {
  int task = blockIdx.x * 8 + (threadIdx.x >> 5);
  int lane = threadIdx.x & 31;
  int bh = task >> 6, rem = task & 63;
  int mt = rem >> 1, nt = rem & 1;
  const f16* A = S + (size_t)bh * kN * kN + (size_t)mt * 16 * kN;
  const f16* Bt = vp + (size_t)bh * kHD * kN;
  v8f acc = zero8();
#pragma unroll 4
  for (int kb = 0; kb < kN; kb += 32)
    acc = wmma_f16(load_a_frag(A, kN, kb, lane), load_b_frag(Bt, kN, nt * 16, kb, lane), acc);
  int hi = lane >> 4;
  int b_ = bh >> 2, h_ = bh & 3;
  int col = h_ * kHD + nt * 16 + (lane & 15);
#pragma unroll
  for (int i = 0; i < 8; ++i) {
    int grow = b_ * kN + mt * 16 + i + 8 * hi;
    op[(size_t)grow * kD + col] = (f16)acc[i];
  }
}

// h = LN(h + op @ OW + ob) ; refresh hp
__global__ __launch_bounds__(256) void k_oproj(const f16* __restrict__ op, const f16* __restrict__ wT,
                                               const float* __restrict__ bias, const float* __restrict__ g,
                                               const float* __restrict__ bb, float* __restrict__ h,
                                               f16* __restrict__ hp) {
  __shared__ float tb[16][kD];
  __shared__ float red0[16][16], red1[16][16];
  __shared__ float smu[16], srs[16];
  int mt = blockIdx.x, wv = threadIdx.x >> 5, lane = threadIdx.x & 31;
  const f16* A = op + (size_t)mt * 16 * kD;
  v8f acc = zero8();
#pragma unroll
  for (int kb = 0; kb < kD; kb += 32)
    acc = wmma_f16(load_a_frag(A, kD, kb, lane), load_b_frag(wT, kD, wv * 16, kb, lane), acc);
  int hi = lane >> 4, col = wv * 16 + (lane & 15);
#pragma unroll
  for (int i = 0; i < 8; ++i) {
    int r = i + 8 * hi;
    tb[r][col] = acc[i] + bias[col] + h[(size_t)(mt * 16 + r) * kD + col];
  }
  __syncthreads();
  int row = threadIdx.x >> 4, seg = threadIdx.x & 15;
  float s = 0.f, s2 = 0.f;
#pragma unroll
  for (int j = 0; j < 8; ++j) { float v = tb[row][seg * 8 + j]; s += v; s2 += v * v; }
  red0[row][seg] = s; red1[row][seg] = s2;
  __syncthreads();
  if (seg == 0) {
    float ts = 0.f, ts2 = 0.f;
#pragma unroll
    for (int j = 0; j < 16; ++j) { ts += red0[row][j]; ts2 += red1[row][j]; }
    float m = ts / kD;
    smu[row] = m;
    srs[row] = rsqrtf(ts2 / kD - m * m + 1e-5f);
  }
  __syncthreads();
#pragma unroll
  for (int j = 0; j < 8; ++j) {
    int c = seg * 8 + j;
    float v = (tb[row][c] - smu[row]) * srs[row] * g[c] + bb[c];
    size_t idx = (size_t)(mt * 16 + row) * kD + c;
    h[idx] = v;
    hp[idx] = (f16)v;
  }
}

// ---------------- pooling + graph head ----------------

__global__ __launch_bounds__(256) void k_pool(const float* __restrict__ h, float* __restrict__ pool) {
  __shared__ float ss[256], sq[256], sm[256];
  int b = blockIdx.x, t = threadIdx.x;
  int c = t & 127, part = t >> 7;
  const float* hb = h + (size_t)b * kN * kD;
  float s = 0.f, q = 0.f, mx = -1e30f;
  for (int n = part * 256; n < part * 256 + 256; ++n) {
    float v = hb[(size_t)n * kD + c];
    s += v; q += v * v; mx = fmaxf(mx, v);
  }
  ss[t] = s; sq[t] = q; sm[t] = mx;
  __syncthreads();
  if (t < 128) {
    float S = ss[t] + ss[t + 128];
    float Q = sq[t] + sq[t + 128];
    float M = fmaxf(sm[t], sm[t + 128]);
    float mean = S / kN;
    float st = sqrtf(fmaxf(Q / kN - mean * mean, 0.f) + 1e-6f);
    float* pb = pool + (size_t)b * 4 * kD;
    pb[c] = mean; pb[kD + c] = S; pb[2 * kD + c] = M; pb[3 * kD + c] = st;
  }
}

__global__ __launch_bounds__(256) void k_head1(const float* __restrict__ pool, const float* __restrict__ w1,
                                               const float* __restrict__ b1, const float* __restrict__ g,
                                               const float* __restrict__ bb, float* __restrict__ ghid) {
  __shared__ float r0[256], r1[256];
  int b = blockIdx.x, t = threadIdx.x;
  const float* pb = pool + (size_t)b * 4 * kD;
  float acc = b1[t];
  for (int k2 = 0; k2 < 4 * kD; ++k2) acc += pb[k2] * w1[(size_t)k2 * kHid + t];
  float v = gelu_f(acc);
  r0[t] = v; r1[t] = v * v;
  __syncthreads();
  for (int off = 128; off; off >>= 1) {
    if (t < off) { r0[t] += r0[t + off]; r1[t] += r1[t + off]; }
    __syncthreads();
  }
  float m = r0[0] / kHid;
  float rs = rsqrtf(r1[0] / kHid - m * m + 1e-5f);
  ghid[(size_t)b * kHid + t] = (v - m) * rs * g[t] + bb[t];
}

__global__ __launch_bounds__(320) void k_head2(const float* __restrict__ ghid, const float* __restrict__ w2,
                                               const float* __restrict__ b2, float* __restrict__ out) {
  int t = threadIdx.x;
  int b = t / kC, c = t % kC;
  const float* gb = ghid + (size_t)b * kHid;
  float acc = b2[c];
  for (int k2 = 0; k2 < kHid; ++k2) acc += gb[k2] * w2[(size_t)k2 * kC + c];
  out[b * kC + c] = acc;
}

// ---------------- launch ----------------

extern "C" void kernel_launch(void* const* d_in, const int* in_sizes, int n_in,
                              void* d_out, int out_size, void* d_ws, size_t ws_size,
                              hipStream_t stream) {
  (void)in_sizes; (void)n_in; (void)out_size; (void)ws_size;
  const float* x      = (const float*)d_in[0];
  const int*   src    = (const int*)d_in[1];
  const int*   dst    = (const int*)d_in[2];
  const float* enc_w1 = (const float*)d_in[3];
  const float* enc_b1 = (const float*)d_in[4];
  const float* enc_g1 = (const float*)d_in[5];
  const float* enc_bn1= (const float*)d_in[6];
  const float* enc_w2 = (const float*)d_in[7];
  const float* enc_b2 = (const float*)d_in[8];
  const float* enc_g2 = (const float*)d_in[9];
  const float* enc_bn2= (const float*)d_in[10];
  const float* msg_w1 = (const float*)d_in[11];
  const float* msg_b1 = (const float*)d_in[12];
  const float* msg_w2 = (const float*)d_in[13];
  const float* msg_b2 = (const float*)d_in[14];
  const float* msg_g  = (const float*)d_in[15];
  const float* msg_bb = (const float*)d_in[16];
  const float* qw     = (const float*)d_in[17];
  const float* qb     = (const float*)d_in[18];
  const float* kw     = (const float*)d_in[19];
  const float* kbv    = (const float*)d_in[20];
  const float* vw     = (const float*)d_in[21];
  const float* vb     = (const float*)d_in[22];
  const float* ow     = (const float*)d_in[23];
  const float* ob     = (const float*)d_in[24];
  const float* attn_g = (const float*)d_in[25];
  const float* attn_b = (const float*)d_in[26];
  const float* gh_w1  = (const float*)d_in[27];
  const float* gh_b1  = (const float*)d_in[28];
  const float* gh_g   = (const float*)d_in[29];
  const float* gh_bn  = (const float*)d_in[30];
  const float* gh_w2  = (const float*)d_in[31];
  const float* gh_b2  = (const float*)d_in[32];
  float* out = (float*)d_out;

  uint8_t* base = (uint8_t*)d_ws;
  size_t off = 0;
  auto carve = [&](size_t bytes) -> void* {
    void* r = base + off;
    off += (bytes + 255) & ~(size_t)255;
    return r;
  };
  f16*   xp     = (f16*)carve((size_t)kBN * kIn * 2);
  f16*   up     = (f16*)carve((size_t)kBN * kHid * 2);
  float* h      = (float*)carve((size_t)kBN * kD * 4);
  f16*   hp     = (f16*)carve((size_t)kBN * kD * 2);
  float* accb   = (float*)carve((size_t)kBN * kD * 4);
  f16*   qp     = (f16*)carve((size_t)kB * kH * kN * kHD * 2);
  f16*   kp     = (f16*)carve((size_t)kB * kH * kN * kHD * 2);
  f16*   vp     = (f16*)carve((size_t)kB * kH * kHD * kN * 2);
  f16*   Sb     = (f16*)carve((size_t)kB * kH * kN * kN * 2);   // 64 MB, L2-resident
  f16*   opb    = (f16*)carve((size_t)kBN * kD * 2);
  float* pool   = (float*)carve((size_t)kB * 4 * kD * 4);
  float* ghid   = (float*)carve((size_t)kB * kHid * 4);
  f16*   encw1T = (f16*)carve((size_t)kIn * kHid * 2);
  f16*   encw2T = (f16*)carve((size_t)kHid * kD * 2);
  f16*   msgw1T = (f16*)carve((size_t)kL * 2 * kD * kD * 2);
  f16*   msgw2T = (f16*)carve((size_t)kL * kD * kD * 2);
  f16*   qwT    = (f16*)carve((size_t)kL * kD * kD * 2);
  f16*   kwT    = (f16*)carve((size_t)kL * kD * kD * 2);
  f16*   vwT    = (f16*)carve((size_t)kL * kD * kD * 2);
  f16*   owT    = (f16*)carve((size_t)kL * kD * kD * 2);

  // weight / input packing (f32 -> f16, column-major [N][K] for B-side)
  k_pack_f16<<<256, 256, 0, stream>>>(x, xp, kBN * kIn);
  k_pack_wT<<<64, 256, 0, stream>>>(enc_w1, encw1T, kIn, kHid);
  k_pack_wT<<<64, 256, 0, stream>>>(enc_w2, encw2T, kHid, kD);
  for (int l = 0; l < kL; ++l) {
    k_pack_wT<<<64, 256, 0, stream>>>(msg_w1 + (size_t)l * 2 * kD * kD, msgw1T + (size_t)l * 2 * kD * kD, 2 * kD, kD);
    k_pack_wT<<<64, 256, 0, stream>>>(msg_w2 + (size_t)l * kD * kD, msgw2T + (size_t)l * kD * kD, kD, kD);
    k_pack_wT<<<64, 256, 0, stream>>>(qw + (size_t)l * kD * kD, qwT + (size_t)l * kD * kD, kD, kD);
    k_pack_wT<<<64, 256, 0, stream>>>(kw + (size_t)l * kD * kD, kwT + (size_t)l * kD * kD, kD, kD);
    k_pack_wT<<<64, 256, 0, stream>>>(vw + (size_t)l * kD * kD, vwT + (size_t)l * kD * kD, kD, kD);
    k_pack_wT<<<64, 256, 0, stream>>>(ow + (size_t)l * kD * kD, owT + (size_t)l * kD * kD, kD, kD);
  }

  // node encoder
  k_enc1<<<kBN / 16, 256, 0, stream>>>(xp, encw1T, enc_b1, enc_g1, enc_bn1, up);
  k_enc2<<<kBN / 16, 256, 0, stream>>>(up, encw2T, enc_b2, enc_g2, enc_bn2, h, hp);

  for (int l = 0; l < kL; ++l) {
    // edge messages + segment-sum
    hipMemsetAsync(accb, 0, (size_t)kBN * kD * sizeof(float), stream);
    k_edge<<<kBE / 64, 256, 0, stream>>>(hp, src, dst,
                                         msgw1T + (size_t)l * 2 * kD * kD, msg_b1 + (size_t)l * kD,
                                         msgw2T + (size_t)l * kD * kD, msg_b2 + (size_t)l * kD, accb);
    k_resln<<<kBN, 128, 0, stream>>>(h, accb, msg_g + (size_t)l * kD, msg_bb + (size_t)l * kD, hp);
    // attention
    k_proj<0><<<kBN / 16, 256, 0, stream>>>(hp, qwT + (size_t)l * kD * kD, qb + (size_t)l * kD, qp);
    k_proj<0><<<kBN / 16, 256, 0, stream>>>(hp, kwT + (size_t)l * kD * kD, kbv + (size_t)l * kD, kp);
    k_proj<1><<<kBN / 16, 256, 0, stream>>>(hp, vwT + (size_t)l * kD * kD, vb + (size_t)l * kD, vp);
    k_scores<<<kB * kH * 32 * 32 / 8, 256, 0, stream>>>(qp, kp, Sb);
    k_softmax<<<kB * kH * kN / 8, 256, 0, stream>>>(Sb);
    k_av<<<kB * kH * 32 * 2 / 8, 256, 0, stream>>>(Sb, vp, opb);
    k_oproj<<<kBN / 16, 256, 0, stream>>>(opb, owT + (size_t)l * kD * kD, ob + (size_t)l * kD,
                                          attn_g + (size_t)l * kD, attn_b + (size_t)l * kD, h, hp);
  }

  // pooling + graph head
  k_pool<<<kB, 256, 0, stream>>>(h, pool);
  k_head1<<<kB, 256, 0, stream>>>(pool, gh_w1, gh_b1, gh_g, gh_bn, ghid);
  k_head2<<<1, 320, 0, stream>>>(ghid, gh_w2, gh_b2, out);
}
